// TokenPruningLayer_64175401337516
// MI455X (gfx1250) — matrix operand order repaired and checked by reading, not compiled, not run on hardware
//
#include <hip/hip_runtime.h>

// ---------------------------------------------------------------- constants
#define B_    2
#define S_    2048
#define D_    1024
#define H_    16
#define F_    4096
#define KEEPK 1638                 // int(2048 * 0.8)
#define MTOT  (B_ * KEEPK)         // 3276
#define MPAD  3328                 // MTOT rounded to 64
#define KPAD  1664                 // KEEPK rounded to 32
#define NQT   103                  // ceil(KEEPK / 16)
#define LN_EPS 1e-5f

typedef __bf16 bf16;
typedef __attribute__((ext_vector_type(8)))  bf16  v8bf;
typedef __attribute__((ext_vector_type(16))) bf16  v16bf;
typedef __attribute__((ext_vector_type(8)))  float v8f;

__device__ __forceinline__ v8f vzero8() {
  v8f z;
#pragma unroll
  for (int i = 0; i < 8; ++i) z[i] = 0.f;
  return z;
}

// Two contiguous 16-byte chunks -> one 16-element bf16 fragment register set.
__device__ __forceinline__ v16bf ld_frag(const bf16* p0, const bf16* p1) {
  v8bf lo = *(const v8bf*)p0;
  v8bf hi = *(const v8bf*)p1;
  return __builtin_shufflevector(lo, hi, 0, 1, 2, 3, 4, 5, 6, 7,
                                 8, 9, 10, 11, 12, 13, 14, 15);
}

__device__ __forceinline__ v8f wmma_bf16(v16bf a, v16bf b, v8f c) {
  // D = A(16x32 bf16) * B(32x16 bf16) + C(16x16 f32)
  return __builtin_amdgcn_wmma_f32_16x16x32_bf16(false, a, false, b,
                                                 (short)0, c, false, false);
}

// ---------------------------------------------------------------- elementwise
__global__ void copy_f32_k(const float* __restrict__ in, float* __restrict__ out, int n) {
  int i = blockIdx.x * 256 + threadIdx.x;
  if (i < n) out[i] = in[i];
}

__global__ __launch_bounds__(256) void token_scores_k(const float* __restrict__ x,
                                                      float* __restrict__ scores) {
  int tok = blockIdx.x;                       // B_*S_ blocks
  const float* row = x + (size_t)tok * D_;
  float s = 0.f;
  for (int i = threadIdx.x; i < D_; i += 256) { float v = row[i]; s += v * v; }
  __shared__ float red[256];
  red[threadIdx.x] = s;
  __syncthreads();
  for (int st = 128; st > 0; st >>= 1) {
    if (threadIdx.x < st) red[threadIdx.x] += red[threadIdx.x + st];
    __syncthreads();
  }
  if (threadIdx.x == 0) scores[tok] = red[0];
}

// Rank-based top-k (stable, matches jax top_k tie-break on lower index) +
// order-preserving compaction into gidx[b*KEEPK + pos].
__global__ __launch_bounds__(1024) void select_topk_k(const float* __restrict__ scores,
                                                      int* __restrict__ gidx) {
  int b = blockIdx.x;
  __shared__ float sc[S_];
  __shared__ int   kept[S_];
  for (int i = threadIdx.x; i < S_; i += 1024) sc[i] = scores[b * S_ + i];
  __syncthreads();
  for (int i = threadIdx.x; i < S_; i += 1024) {
    float s = sc[i];
    int rank = 0;
    for (int j = 0; j < S_; ++j) {
      float sj = sc[j];
      rank += (sj > s) || (sj == s && j < i);
    }
    kept[i] = (rank < KEEPK) ? 1 : 0;
  }
  __syncthreads();
  for (int i = threadIdx.x; i < S_; i += 1024) {
    if (kept[i]) {
      int pos = 0;
      for (int j = 0; j < i; ++j) pos += kept[j];
      gidx[b * KEEPK + pos] = i;
    }
  }
}

__global__ void gather_k(const float* __restrict__ hid, const int* __restrict__ gidx,
                         float* __restrict__ pruned, int n) {
  int i = blockIdx.x * 256 + threadIdx.x;
  if (i >= n) return;
  int row = i / D_, d = i % D_;
  int b = row / KEEPK;
  int tok = gidx[row];
  pruned[(size_t)row * D_ + d] = hid[((size_t)b * S_ + tok) * (size_t)D_ + d];
}

__global__ __launch_bounds__(256) void layernorm_k(const float* __restrict__ x,
                                                   const float* __restrict__ sc,
                                                   const float* __restrict__ bi,
                                                   bf16* __restrict__ out) {
  int row = blockIdx.x;
  const float* xr = x + (size_t)row * D_;
  float s1 = 0.f, s2 = 0.f;
  for (int i = threadIdx.x; i < D_; i += 256) { float v = xr[i]; s1 += v; s2 += v * v; }
  __shared__ float r1[256], r2[256];
  r1[threadIdx.x] = s1;
  r2[threadIdx.x] = s2;
  __syncthreads();
  for (int st = 128; st > 0; st >>= 1) {
    if (threadIdx.x < st) {
      r1[threadIdx.x] += r1[threadIdx.x + st];
      r2[threadIdx.x] += r2[threadIdx.x + st];
    }
    __syncthreads();
  }
  float mean = r1[0] * (1.f / D_);
  float var  = r2[0] * (1.f / D_) - mean * mean;
  float rstd = rsqrtf(var + LN_EPS);
  bf16* orow = out + (size_t)row * D_;
  for (int i = threadIdx.x; i < D_; i += 256)
    orow[i] = (bf16)((xr[i] - mean) * rstd * sc[i] + bi[i]);
}

// fp32 W[K,N] -> bf16 WT[N,K]  (transposed so B-fragments are contiguous)
__global__ void wconv_k(const float* __restrict__ W, bf16* __restrict__ WT,
                        int K, int N, int n) {
  int i = blockIdx.x * 256 + threadIdx.x;
  if (i >= n) return;
  int k = i / N, col = i % N;
  WT[(size_t)col * K + k] = (bf16)W[i];
}

// v bf16 [MPAD, D] -> vT bf16 [B, H, 64, KPAD], zero-filling key pad
__global__ void vtrans_k(const bf16* __restrict__ v, bf16* __restrict__ vT, int n) {
  int i = blockIdx.x * 256 + threadIdx.x;
  if (i >= n) return;
  int r  = i % KPAD;
  int dh = (i / KPAD) % 64;
  int h  = (i / (KPAD * 64)) % H_;
  int b  = i / (KPAD * 64 * H_);
  bf16 val = (bf16)0.f;
  if (r < KEEPK) val = v[(size_t)(b * KEEPK + r) * D_ + h * 64 + dh];
  vT[i] = val;
}

__global__ void scatter_k(const float* __restrict__ y, const int* __restrict__ gidx,
                          float* __restrict__ out, int n) {
  int i = blockIdx.x * 256 + threadIdx.x;
  if (i >= n) return;
  int row = i / D_, d = i % D_;
  int b = row / KEEPK;
  int tok = gidx[row];
  out[((size_t)b * S_ + tok) * (size_t)D_ + d] = y[(size_t)row * D_ + d];
}

// ---------------------------------------------------------------- WMMA GEMM
// C[M,N] = A[M,K](bf16,row-major) * WT[N,K](bf16)^T + bias, epilogue by mode:
//   mode 0: bf16 out = C + bias
//   mode 1: f32  out = C + bias + res
//   mode 2: bf16 out = gelu_tanh(C + bias)
// Block: 256 thr = 8 waves; wave(wm,wn) owns 16x64 C strip; block tile 64x128.
// The K loop is ping-pong double-buffered with BOTH sets preloaded, so every
// fragment load issues a full half-iteration before its consuming WMMA:
//   steady state: WMMA(set0,k); load set0<-k+64; WMMA(set1,k+32); load set1<-k+96
// K/32 is even for all call sites (K = 1024 or 4096).
__global__ __launch_bounds__(256) void gemm_k(const bf16* __restrict__ A,
                                              const bf16* __restrict__ WT,
                                              const float* __restrict__ bias,
                                              const float* __restrict__ res,
                                              void* __restrict__ out,
                                              int M, int N, int K, int mode) {
  int tid = threadIdx.x;
  int lane = tid & 31;
  int w = tid >> 5;
  int wm = w & 3, wn = w >> 2;
  int m0 = blockIdx.y * 64 + wm * 16;
  int n0 = blockIdx.x * 128 + wn * 64;
  int l15 = lane & 15;
  int kOffA = (lane >> 4) * 16;   // A frag: K 0-15 (lanes 0-15) or 16-31
  int kOffB = (lane >> 4) * 16;   // B frag: K 0-15 (lanes 0-15) or 16-31
  const bf16* ap  = A  + (size_t)(m0 + l15) * K + kOffA;
  const bf16* bp0 = WT + (size_t)(n0 + l15) * K + kOffB;
  const bf16* bp1 = bp0 + (size_t)16 * K;
  const bf16* bp2 = bp0 + (size_t)32 * K;
  const bf16* bp3 = bp0 + (size_t)48 * K;

  v8f c[4];
#pragma unroll
  for (int j = 0; j < 4; ++j) c[j] = vzero8();

  v16bf a0, a1, b0[4], b1[4];
  // prologue: set 0 <- k=0, set 1 <- k=32
  a0 = ld_frag(ap, ap + 8);
  b0[0] = ld_frag(bp0, bp0 + 8);
  b0[1] = ld_frag(bp1, bp1 + 8);
  b0[2] = ld_frag(bp2, bp2 + 8);
  b0[3] = ld_frag(bp3, bp3 + 8);
  a1 = ld_frag(ap + 32, ap + 40);
  b1[0] = ld_frag(bp0 + 32, bp0 + 40);
  b1[1] = ld_frag(bp1 + 32, bp1 + 40);
  b1[2] = ld_frag(bp2 + 32, bp2 + 40);
  b1[3] = ld_frag(bp3 + 32, bp3 + 40);

  int k0 = 0;
  for (; k0 + 128 <= K; k0 += 64) {
    int ka = k0 + 64, kb = k0 + 96;
    __builtin_prefetch((const void*)(ap + k0 + 512), 0, 1);
    // consume set 0 (k0), refill it with k0+64
#pragma unroll
    for (int j = 0; j < 4; ++j) c[j] = wmma_bf16(a0, b0[j], c[j]);
    a0 = ld_frag(ap + ka, ap + ka + 8);
    b0[0] = ld_frag(bp0 + ka, bp0 + ka + 8);
    b0[1] = ld_frag(bp1 + ka, bp1 + ka + 8);
    b0[2] = ld_frag(bp2 + ka, bp2 + ka + 8);
    b0[3] = ld_frag(bp3 + ka, bp3 + ka + 8);
    // consume set 1 (k0+32), refill it with k0+96
#pragma unroll
    for (int j = 0; j < 4; ++j) c[j] = wmma_bf16(a1, b1[j], c[j]);
    a1 = ld_frag(ap + kb, ap + kb + 8);
    b1[0] = ld_frag(bp0 + kb, bp0 + kb + 8);
    b1[1] = ld_frag(bp1 + kb, bp1 + kb + 8);
    b1[2] = ld_frag(bp2 + kb, bp2 + kb + 8);
    b1[3] = ld_frag(bp3 + kb, bp3 + kb + 8);
  }
  // tail: sets hold K-64 and K-32
#pragma unroll
  for (int j = 0; j < 4; ++j) c[j] = wmma_bf16(a0, b0[j], c[j]);
#pragma unroll
  for (int j = 0; j < 4; ++j) c[j] = wmma_bf16(a1, b1[j], c[j]);

  int rOff = (lane >> 4) * 8;     // C frag rows: vgpr r -> row r (+8 for hi lanes)
#pragma unroll
  for (int j = 0; j < 4; ++j) {
    int col = n0 + j * 16 + l15;
    float bv = bias[col];
#pragma unroll
    for (int r = 0; r < 8; ++r) {
      int row = m0 + rOff + r;
      if (row < M) {
        float v = c[j][r] + bv;
        if (mode == 0) {
          ((bf16*)out)[(size_t)row * N + col] = (bf16)v;
        } else if (mode == 1) {
          ((float*)out)[(size_t)row * N + col] = v + res[(size_t)row * N + col];
        } else {
          float t = 0.7978845608028654f * (v + 0.044715f * v * v * v);
          float g = 0.5f * v * (1.f + tanhf(t));
          ((bf16*)out)[(size_t)row * N + col] = (bf16)g;
        }
      }
    }
  }
}

// ---------------------------------------------------------------- attention
// One wave per (b, h, 16-query tile). Flash-style over 32-key chunks:
// S = Q Kh^T / 8 (2 WMMA per 16-key tile), online softmax (shfl reductions in
// 16-lane halves match the C-fragment row layout), P staged via LDS to
// A-fragment layout, O accumulated with 4 WMMAs against pre-transposed V.
// Pipelining: V fragments load at chunk top; K fragments are ping-pong
// double-buffered (52 chunks = even), next chunk's K loads issue right after
// the S WMMAs so their latency hides behind softmax + LDS staging.
__global__ __launch_bounds__(32) void attn_k(const bf16* __restrict__ Q,
                                             const bf16* __restrict__ Kb,
                                             const bf16* __restrict__ vT,
                                             bf16* __restrict__ O) {
  int lane = threadIdx.x & 31;
  int l15 = lane & 15;
  int hi16 = lane >> 4;
  int bid = blockIdx.x;
  int qt = bid % NQT;
  int h  = (bid / NQT) % H_;
  int b  = bid / (NQT * H_);
  int q0 = qt * 16;
  size_t rowbase = (size_t)b * KEEPK;
  int kOffA = hi16 * 16;
  int kOffB = hi16 * 16;

  const bf16* qp = Q + (rowbase + q0 + l15) * D_ + h * 64;
  v16bf qa0 = ld_frag(qp + kOffA, qp + kOffA + 8);           // dh 0..31
  v16bf qa1 = ld_frag(qp + 32 + kOffA, qp + 32 + kOffA + 8); // dh 32..63

  v8f o0 = vzero8(), o1 = vzero8(), o2 = vzero8(), o3 = vzero8();
  float m[8], l[8];
#pragma unroll
  for (int r = 0; r < 8; ++r) { m[r] = -1e30f; l[r] = 0.f; }

  __shared__ __align__(16) bf16 pbuf[16 * 32];
  const bf16* vbase = vT + ((size_t)(b * H_ + h) * 64) * KPAD;

  auto load_kfrag = [&](int kc, v16bf* kb) {
    const bf16* kp0 = Kb + (rowbase + kc + l15) * D_ + h * 64 + kOffB;
    kb[0] = ld_frag(kp0, kp0 + 8);
    kb[1] = ld_frag(kp0 + 32, kp0 + 40);
    const bf16* kp1 = kp0 + (size_t)16 * D_;
    kb[2] = ld_frag(kp1, kp1 + 8);
    kb[3] = ld_frag(kp1 + 32, kp1 + 40);
  };

  // Process one 32-key chunk using fragments kb; issue next chunk's K loads
  // (into kbn) immediately after the S WMMAs consume kb.
  auto process_chunk = [&](int kc, v16bf* kb, int kcn, v16bf* kbn) {
    // V fragments for this chunk: independent of S, issue early.
    v16bf vbf[4];
#pragma unroll
    for (int d = 0; d < 4; ++d) {
      const bf16* vp = vbase + (size_t)(d * 16 + l15) * KPAD + kc + kOffB;
      vbf[d] = ld_frag(vp, vp + 8);
    }
    // S tiles from current K fragments.
    v8f s0 = wmma_bf16(qa0, kb[0], vzero8());
    s0 = wmma_bf16(qa1, kb[1], s0);
    v8f s1 = wmma_bf16(qa0, kb[2], vzero8());
    s1 = wmma_bf16(qa1, kb[3], s1);
    // Next chunk's K fragments into the other buffer.
    load_kfrag(kcn, kbn);

    const float sc = 0.125f;   // 1/sqrt(64)
    bool mask0 = (kc + l15) >= KEEPK;
    bool mask1 = (kc + 16 + l15) >= KEEPK;
#pragma unroll
    for (int r = 0; r < 8; ++r) {
      s0[r] = mask0 ? -1e30f : s0[r] * sc;
      s1[r] = mask1 ? -1e30f : s1[r] * sc;
    }
    float mx[8];
#pragma unroll
    for (int r = 0; r < 8; ++r) mx[r] = fmaxf(s0[r], s1[r]);
#pragma unroll
    for (int off = 1; off < 16; off <<= 1)
#pragma unroll
      for (int r = 0; r < 8; ++r) mx[r] = fmaxf(mx[r], __shfl_xor(mx[r], off, 32));
    float scl[8];
#pragma unroll
    for (int r = 0; r < 8; ++r) {
      float mn = fmaxf(m[r], mx[r]);
      scl[r] = __expf(m[r] - mn);
      m[r] = mn;
    }
#pragma unroll
    for (int r = 0; r < 8; ++r) {
      s0[r] = __expf(s0[r] - m[r]);
      s1[r] = __expf(s1[r] - m[r]);
    }
    float sum[8];
#pragma unroll
    for (int r = 0; r < 8; ++r) sum[r] = s0[r] + s1[r];
#pragma unroll
    for (int off = 1; off < 16; off <<= 1)
#pragma unroll
      for (int r = 0; r < 8; ++r) sum[r] += __shfl_xor(sum[r], off, 32);
#pragma unroll
    for (int r = 0; r < 8; ++r) l[r] = l[r] * scl[r] + sum[r];
#pragma unroll
    for (int r = 0; r < 8; ++r) {
      o0[r] *= scl[r]; o1[r] *= scl[r]; o2[r] *= scl[r]; o3[r] *= scl[r];
    }
    // P (C-layout) -> LDS -> A-layout fragment
    int rBase = hi16 * 8;
#pragma unroll
    for (int r = 0; r < 8; ++r) {
      pbuf[(rBase + r) * 32 + l15]      = (bf16)s0[r];
      pbuf[(rBase + r) * 32 + 16 + l15] = (bf16)s1[r];
    }
    __syncthreads();
    const bf16* pp = &pbuf[l15 * 32 + kOffA];
    v16bf pa = ld_frag(pp, pp + 8);
    o0 = wmma_bf16(pa, vbf[0], o0);
    o1 = wmma_bf16(pa, vbf[1], o1);
    o2 = wmma_bf16(pa, vbf[2], o2);
    o3 = wmma_bf16(pa, vbf[3], o3);
    __syncthreads();
  };

  v16bf kb0[4], kb1[4];
  load_kfrag(0, kb0);
  // 52 chunks of 32 keys, processed two per iteration (ping-pong, no copies).
  for (int kc = 0; kc < KEEPK; kc += 64) {
    process_chunk(kc, kb0, kc + 32, kb1);
    int kn = kc + 64;
    process_chunk(kc + 32, kb1, (kn < KEEPK) ? kn : 0, kb0);
  }

  int rBase = hi16 * 8;
#pragma unroll
  for (int r = 0; r < 8; ++r) {
    int qrow = q0 + rBase + r;
    if (qrow < KEEPK) {
      float inv = 1.f / l[r];
      bf16* op = O + (rowbase + qrow) * D_ + h * 64;
      op[l15]      = (bf16)(o0[r] * inv);
      op[16 + l15] = (bf16)(o1[r] * inv);
      op[32 + l15] = (bf16)(o2[r] * inv);
      op[48 + l15] = (bf16)(o3[r] * inv);
    }
  }
}

// ---------------------------------------------------------------- launcher
extern "C" void kernel_launch(void* const* d_in, const int* in_sizes, int n_in,
                              void* d_out, int out_size, void* d_ws, size_t ws_size,
                              hipStream_t stream) {
  (void)in_sizes; (void)n_in; (void)out_size; (void)ws_size;
  const float* hid  = (const float*)d_in[0];
  const float* ln1s = (const float*)d_in[1];
  const float* ln1b = (const float*)d_in[2];
  const float* wq   = (const float*)d_in[3];
  const float* bq   = (const float*)d_in[4];
  const float* wk   = (const float*)d_in[5];
  const float* bk   = (const float*)d_in[6];
  const float* wv   = (const float*)d_in[7];
  const float* bv   = (const float*)d_in[8];
  const float* wo   = (const float*)d_in[9];
  const float* bo   = (const float*)d_in[10];
  const float* ln2s = (const float*)d_in[11];
  const float* ln2b = (const float*)d_in[12];
  const float* w1   = (const float*)d_in[13];
  const float* b1   = (const float*)d_in[14];
  const float* w2   = (const float*)d_in[15];
  const float* b2   = (const float*)d_in[16];
  float* out = (float*)d_out;

  char* p = (char*)d_ws;
  auto alloc = [&](size_t bytes) -> void* {
    void* r = (void*)p;
    p += (bytes + 255) & ~(size_t)255;
    return r;
  };
  float* scores = (float*)alloc((size_t)B_ * S_ * 4);
  int*   gidx   = (int*)  alloc((size_t)MTOT * 4);
  float* pruned = (float*)alloc((size_t)MPAD * D_ * 4);
  bf16*  hA     = (bf16*) alloc((size_t)MPAD * D_ * 2);
  bf16*  wqT    = (bf16*) alloc((size_t)D_ * D_ * 2);
  bf16*  wkT    = (bf16*) alloc((size_t)D_ * D_ * 2);
  bf16*  wvT    = (bf16*) alloc((size_t)D_ * D_ * 2);
  bf16*  woT    = (bf16*) alloc((size_t)D_ * D_ * 2);
  bf16*  w1T    = (bf16*) alloc((size_t)D_ * F_ * 2);
  bf16*  w2T    = (bf16*) alloc((size_t)F_ * D_ * 2);
  bf16*  qb     = (bf16*) alloc((size_t)MPAD * D_ * 2);
  bf16*  kb     = (bf16*) alloc((size_t)MPAD * D_ * 2);
  bf16*  vb     = (bf16*) alloc((size_t)MPAD * D_ * 2);
  bf16*  vTb    = (bf16*) alloc((size_t)B_ * H_ * 64 * KPAD * 2);
  bf16*  attnb  = (bf16*) alloc((size_t)MPAD * D_ * 2);
  float* x1     = (float*)alloc((size_t)MPAD * D_ * 4);
  bf16*  h2b    = (bf16*) alloc((size_t)MPAD * D_ * 2);
  bf16*  gb     = (bf16*) alloc((size_t)MPAD * F_ * 2);
  float* yb     = (float*)alloc((size_t)MPAD * D_ * 4);

  // 1) pass-through copy: unselected tokens keep their values
  int nAll = B_ * S_ * D_;
  copy_f32_k<<<(nAll + 255) / 256, 256, 0, stream>>>(hid, out, nAll);
  // 2) token selection
  token_scores_k<<<B_ * S_, 256, 0, stream>>>(hid, scores);
  select_topk_k<<<B_, 1024, 0, stream>>>(scores, gidx);
  int nPr = MTOT * D_;
  gather_k<<<(nPr + 255) / 256, 256, 0, stream>>>(hid, gidx, pruned, nPr);
  // 3) LN1 -> bf16 activations
  layernorm_k<<<MTOT, 256, 0, stream>>>(pruned, ln1s, ln1b, hA);
  // 4) weight convert+transpose to bf16 [N,K]
  int nW = D_ * D_;
  wconv_k<<<(nW + 255) / 256, 256, 0, stream>>>(wq, wqT, D_, D_, nW);
  wconv_k<<<(nW + 255) / 256, 256, 0, stream>>>(wk, wkT, D_, D_, nW);
  wconv_k<<<(nW + 255) / 256, 256, 0, stream>>>(wv, wvT, D_, D_, nW);
  wconv_k<<<(nW + 255) / 256, 256, 0, stream>>>(wo, woT, D_, D_, nW);
  int nWm = D_ * F_;
  wconv_k<<<(nWm + 255) / 256, 256, 0, stream>>>(w1, w1T, D_, F_, nWm); // [D,F]->[F,D]
  wconv_k<<<(nWm + 255) / 256, 256, 0, stream>>>(w2, w2T, F_, D_, nWm); // [F,D]->[D,F]
  // 5) QKV projections
  dim3 gProj(D_ / 128, MPAD / 64);
  gemm_k<<<gProj, 256, 0, stream>>>(hA, wqT, bq, nullptr, qb, MTOT, D_, D_, 0);
  gemm_k<<<gProj, 256, 0, stream>>>(hA, wkT, bk, nullptr, kb, MTOT, D_, D_, 0);
  gemm_k<<<gProj, 256, 0, stream>>>(hA, wvT, bv, nullptr, vb, MTOT, D_, D_, 0);
  // 6) attention
  int nVT = B_ * H_ * 64 * KPAD;
  vtrans_k<<<(nVT + 255) / 256, 256, 0, stream>>>(vb, vTb, nVT);
  attn_k<<<B_ * H_ * NQT, 32, 0, stream>>>(qb, kb, vTb, attnb);
  // 7) output projection + residual -> x1 (f32)
  gemm_k<<<gProj, 256, 0, stream>>>(attnb, woT, bo, pruned, x1, MTOT, D_, D_, 1);
  // 8) LN2 + MLP with GELU + residual
  layernorm_k<<<MTOT, 256, 0, stream>>>(x1, ln2s, ln2b, h2b);
  dim3 gMlp1(F_ / 128, MPAD / 64);
  gemm_k<<<gMlp1, 256, 0, stream>>>(h2b, w1T, b1, nullptr, gb, MTOT, F_, D_, 2);
  gemm_k<<<gProj, 256, 0, stream>>>(gb, w2T, b2, x1, yb, MTOT, D_, F_, 1);
  // 9) scatter selected tokens back
  scatter_k<<<(nPr + 255) / 256, 256, 0, stream>>>(yb, gidx, out, nPr);
}
